// SoftHd_23733989278210
// MI455X (gfx1250) — matrix operand order repaired and checked by reading, not compiled, not run on hardware
//
#include <hip/hip_runtime.h>
#include <hip/hip_bf16.h>

// SoftHd on MI455X (gfx1250, wave32):
//  - per (b1,b2) block: 128x128x128 GEMM via v_wmma_f32_16x16x32_bf16,
//    f32 inputs split hi/lo bf16 (3-term emulation ~ f32 accuracy),
//    3 independent accumulators so the 12 WMMAs per tile form three short
//    independent D->C chains instead of one serial 12-deep chain.
//  - fused sqrt/min epilogue, LDS u32-min atomics (bitwise == float order for >=0),
//    deterministic tree-sum.

typedef __attribute__((ext_vector_type(16))) __bf16        v16bf;
typedef __attribute__((ext_vector_type(8)))  float         v8f;
typedef __attribute__((ext_vector_type(8)))  unsigned int  v8u;

#define N_NODES 128
#define DDIM    128
#define LSTRIDE 136   // 128 + 8 pad bf16 elems -> 272B pitch (68 banks): conflict-free b128 frag loads

__device__ __forceinline__ unsigned short f32_to_bf16_rne(float f) {
  unsigned u = __float_as_uint(f);
  u += 0x7FFFu + ((u >> 16) & 1u);        // round-to-nearest-even
  return (unsigned short)(u >> 16);
}

__device__ __forceinline__ v16bf ld_frag32B(const unsigned short* p0, const unsigned short* p1) {
  const uint4 c0 = *(const uint4*)p0;     // ds_load_b128
  const uint4 c1 = *(const uint4*)p1;     // ds_load_b128
  v8u u = { c0.x, c0.y, c0.z, c0.w, c1.x, c1.y, c1.z, c1.w };
  return __builtin_bit_cast(v16bf, u);
}

__global__ __launch_bounds__(256) void softhd_wmma_kernel(
    const float* __restrict__ v1, const int* __restrict__ sz1,
    const float* __restrict__ v2, const int* __restrict__ sz2,
    float* __restrict__ out, int B2) {
  __shared__ unsigned short Ahi[N_NODES * LSTRIDE];
  __shared__ unsigned short Alo[N_NODES * LSTRIDE];
  __shared__ unsigned short Bhi[N_NODES * LSTRIDE];
  __shared__ unsigned short Blo[N_NODES * LSTRIDE];
  __shared__ float    x2s[N_NODES], y2s[N_NODES];
  __shared__ unsigned rowMinU[N_NODES], colMinU[N_NODES];
  __shared__ float    red[256];

  const int t  = threadIdx.x;
  const int b2 = blockIdx.x;
  const int b1 = blockIdx.y;
  const int s1 = sz1[b1];
  const int s2 = sz2[b2];

  // ---- Stage: one 128-float row per thread -> bf16 hi/lo in LDS + exact f32 |row|^2
  {
    const int  row = t & 127;
    const bool isA = t < 128;
    const float* src = isA ? (v1 + ((size_t)b1 * N_NODES + row) * DDIM)
                           : (v2 + ((size_t)b2 * N_NODES + row) * DDIM);
    unsigned short* dH = isA ? Ahi : Bhi;
    unsigned short* dL = isA ? Alo : Blo;
    float nrm = 0.f;
    #pragma unroll
    for (int d = 0; d < DDIM; d += 4) {
      const float4 f = *(const float4*)(src + d);     // global_load_b128, L2-hot
      const float fv[4] = { f.x, f.y, f.z, f.w };
      unsigned hx[4], lx[4];
      #pragma unroll
      for (int k = 0; k < 4; ++k) {
        nrm = __builtin_fmaf(fv[k], fv[k], nrm);
        const unsigned short h  = f32_to_bf16_rne(fv[k]);
        const float          hf = __uint_as_float((unsigned)h << 16);
        hx[k] = h;
        lx[k] = f32_to_bf16_rne(fv[k] - hf);
      }
      uint2 ph = { hx[0] | (hx[1] << 16), hx[2] | (hx[3] << 16) };
      uint2 pl = { lx[0] | (lx[1] << 16), lx[2] | (lx[3] << 16) };
      *(uint2*)&dH[row * LSTRIDE + d] = ph;           // ds_store_b64
      *(uint2*)&dL[row * LSTRIDE + d] = pl;
    }
    if (isA) { x2s[row] = nrm; rowMinU[row] = 0x7F800000u; }
    else     { y2s[row] = nrm; colMinU[row] = 0x7F800000u; }
  }
  __syncthreads();

  // ---- WMMA phase: wave w owns rows [16w, 16w+16), all 128 cols (8 tiles of 16x16)
  const int lane    = t & 31;
  const int wv      = t >> 5;
  const int half    = lane >> 4;
  const int lm      = lane & 15;
  const int rowbase = wv * 16;

  // A fragments (16x32 bf16 layout): lane(0-15) row=lm, elems 0-7 -> K=8h+[0,8), elems 8-15 -> K=16+8h+[0,8)
  v16bf aH[4], aL[4];
  #pragma unroll
  for (int kk = 0; kk < 4; ++kk) {
    const int m  = rowbase + lm;
    const int o1 = m * LSTRIDE + kk * 32 + half * 8;
    const int o2 = o1 + 16;
    aH[kk] = ld_frag32B(&Ahi[o1], &Ahi[o2]);
    aL[kk] = ld_frag32B(&Alo[o1], &Alo[o2]);
  }
  float x2r[8];
  #pragma unroll
  for (int r = 0; r < 8; ++r) x2r[r] = x2s[rowbase + half * 8 + r];

  const float INF = __uint_as_float(0x7F800000u);
  float rowMin[8];
  #pragma unroll
  for (int r = 0; r < 8; ++r) rowMin[r] = INF;

  for (int tc = 0; tc < 8; ++tc) {
    const int   j    = tc * 16 + lm;
    const float y2j  = y2s[j];
    const bool  jval = j < s2;
    // Three independent accumulator chains: hi*hi, lo*hi, hi*lo
    v8f accHH = { 0.f, 0.f, 0.f, 0.f, 0.f, 0.f, 0.f, 0.f };
    v8f accLH = { 0.f, 0.f, 0.f, 0.f, 0.f, 0.f, 0.f, 0.f };
    v8f accHL = { 0.f, 0.f, 0.f, 0.f, 0.f, 0.f, 0.f, 0.f };
    #pragma unroll
    for (int kk = 0; kk < 4; ++kk) {
      // B fragment (32x16): lane col n=lm, elems 0-7 -> K=8h+[0,8), elems 8-15 -> K=16+8h+[0,8)
      const int o1 = j * LSTRIDE + kk * 32 + half * 8;
      const int o2 = o1 + 16;
      const v16bf bH = ld_frag32B(&Bhi[o1], &Bhi[o2]);
      const v16bf bL = ld_frag32B(&Blo[o1], &Blo[o2]);
      accHH = __builtin_amdgcn_wmma_f32_16x16x32_bf16(false, aH[kk], false, bH, (short)0, accHH, false, false);
      accLH = __builtin_amdgcn_wmma_f32_16x16x32_bf16(false, aL[kk], false, bH, (short)0, accLH, false, false);
      accHL = __builtin_amdgcn_wmma_f32_16x16x32_bf16(false, aH[kk], false, bL, (short)0, accHL, false, false);
    }
    // Fused epilogue: dist = sqrt(max(|x|^2+|y|^2-2xy, 0)); element r is (M=r+8*half, N=lm)
    float colMin = INF;
    #pragma unroll
    for (int r = 0; r < 8; ++r) {
      const int   i    = rowbase + half * 8 + r;
      const float xy   = accHH[r] + accLH[r] + accHL[r];
      const float sq   = x2r[r] + y2j - 2.f * xy;
      const float dist = __builtin_sqrtf(__builtin_fmaxf(sq, 0.f));
      const bool  ival = i < s1;
      colMin    = __builtin_fminf(colMin,    ival ? dist : INF);
      rowMin[r] = __builtin_fminf(rowMin[r], jval ? dist : INF);
    }
    if (jval) atomicMin(&colMinU[j], __float_as_uint(colMin));   // ds_min_u32 (order-independent)
  }
  #pragma unroll
  for (int r = 0; r < 8; ++r) {
    const int i = rowbase + half * 8 + r;
    if (i < s1) atomicMin(&rowMinU[i], __float_as_uint(rowMin[r]));
  }
  __syncthreads();

  // ---- Deterministic final sum: bm1 over j<sz2 + bm2 over i<sz1
  float s = 0.f;
  if (t < 128) { if (t < s2)       s = __uint_as_float(colMinU[t]); }
  else         { const int i = t - 128; if (i < s1) s = __uint_as_float(rowMinU[i]); }
  red[t] = s;
  __syncthreads();
  #pragma unroll
  for (int off = 128; off > 0; off >>= 1) {
    if (t < off) red[t] += red[t + off];
    __syncthreads();
  }
  if (t == 0) out[(size_t)b1 * B2 + b2] = (s1 > 0 && s2 > 0) ? red[0] : 0.f;
}

extern "C" void kernel_launch(void* const* d_in, const int* in_sizes, int n_in,
                              void* d_out, int out_size, void* d_ws, size_t ws_size,
                              hipStream_t stream) {
  (void)n_in; (void)out_size; (void)d_ws; (void)ws_size;
  const float* v1  = (const float*)d_in[0];
  const int*   sz1 = (const int*)d_in[1];
  const float* v2  = (const float*)d_in[2];
  const int*   sz2 = (const int*)d_in[3];
  const int B1 = in_sizes[1];   // number of graphs in batch 1 (== sz1 length)
  const int B2 = in_sizes[3];   // number of graphs in batch 2
  dim3 grid(B2, B1);
  softhd_wmma_kernel<<<grid, 256, 0, stream>>>(v1, sz1, v2, sz2, (float*)d_out, B2);
}
